// endecoder2_20839181320647
// MI455X (gfx1250) — matrix operand order repaired and checked by reading, not compiled, not run on hardware
//
#include <hip/hip_runtime.h>
#include <math.h>

// ---------------------------------------------------------------------------
// Types for CDNA5 WMMA (wave32): v_wmma_f32_16x16x32_bf16
// ---------------------------------------------------------------------------
typedef __bf16 bf16_t;
typedef __attribute__((ext_vector_type(16))) __bf16 v16bf;
typedef __attribute__((ext_vector_type(8)))  __bf16 v8bf;
typedef __attribute__((ext_vector_type(8)))  float  v8f;

#define DMODEL 512
#define NTOK   1024
#define BATCH  8
#define NHEAD  8
#define DK     64
#define ROWS   (BATCH * NTOK)   // 8192
#define BH     (BATCH * NHEAD)  // 64

// ---------------------------------------------------------------------------
// Fragment loaders: A and B both use contiguous k order (k = 16*(lane/16) + s)
//
// A (16x32 bf16, MxK): lanes 0-15 row M=lane hold K 0..15 contiguous;
//                      lanes 16-31 row M=lane-16 hold K 16..31.
// B (32x16 bf16, KxN): lanes 0-15 col N=lane hold K 0..15 contiguous;
//                      lanes 16-31 col N=lane-16 hold K 16..31.
// C/D (16x16 f32):     element (m,n): vgpr = m%8, lane = (m/8)*16 + n.
// ---------------------------------------------------------------------------
static __device__ __forceinline__ v16bf load_a_frag(const bf16_t* __restrict__ A,
                                                    int lda, int m0, int k0, int lane) {
  int m  = m0 + (lane & 15);
  int kb = k0 + ((lane >> 4) << 4);           // +16 for upper half-wave
  const bf16_t* p = A + (size_t)m * lda + kb;
  v8bf lo = *(const v8bf*)(p);
  v8bf hi = *(const v8bf*)(p + 8);
  v16bf a;
#pragma unroll
  for (int i = 0; i < 8; ++i) { a[i] = lo[i]; a[i + 8] = hi[i]; }
  return a;
}

static __device__ __forceinline__ v16bf load_b_frag(const bf16_t* __restrict__ Bt,
                                                    int ldb, int n0, int k0, int lane) {
  int n  = n0 + (lane & 15);
  int kb = k0 + ((lane >> 4) << 4);           // +16 for upper half-wave
  return *(const v16bf*)(Bt + (size_t)n * ldb + kb);
}

// One wave computes a 32x64 tile: 2 A fragments x 4 B fragments -> 8 WMMAs per
// K-step on 12 b128 loads (B fragments amortized over two M-tiles).
// A may be split in K (A1 for k<K1, A2 for k>=K1) for the concat([r,d]) GEMMs.
static __device__ __forceinline__ void gemm_tile(const bf16_t* __restrict__ A1,
                                                 const bf16_t* __restrict__ A2,
                                                 int K1, int K, int lda,
                                                 const bf16_t* __restrict__ Bt, int ldb,
                                                 int m0, int n0, int lane, v8f acc[2][4]) {
  for (int k0 = 0; k0 < K; k0 += 32) {
    v16bf a0, a1;
    if (k0 < K1) {
      a0 = load_a_frag(A1, lda, m0,      k0, lane);
      a1 = load_a_frag(A1, lda, m0 + 16, k0, lane);
    } else {
      a0 = load_a_frag(A2, lda, m0,      k0 - K1, lane);
      a1 = load_a_frag(A2, lda, m0 + 16, k0 - K1, lane);
    }
#pragma unroll
    for (int t = 0; t < 4; ++t) {
      v16bf b = load_b_frag(Bt, ldb, n0 + t * 16, k0, lane);
      acc[0][t] = __builtin_amdgcn_wmma_f32_16x16x32_bf16(
          false, a0, false, b, (short)0, acc[0][t], false, false);
      acc[1][t] = __builtin_amdgcn_wmma_f32_16x16x32_bf16(
          false, a1, false, b, (short)0, acc[1][t], false, false);
    }
  }
}

#define ZERO_ACC(acc) \
  v8f acc[2][4]; { v8f z_ = {0.f,0.f,0.f,0.f,0.f,0.f,0.f,0.f}; \
    acc[0][0] = z_; acc[0][1] = z_; acc[0][2] = z_; acc[0][3] = z_; \
    acc[1][0] = z_; acc[1][1] = z_; acc[1][2] = z_; acc[1][3] = z_; }

// ---------------------------------------------------------------------------
// Weight convert + transpose: W[K][N] f32 -> Wt[N][K] bf16 (K contiguous)
// ---------------------------------------------------------------------------
__global__ void wt_kernel(const float* __restrict__ W, bf16_t* __restrict__ Wt,
                          int K, int N) {
  int idx = blockIdx.x * 256 + threadIdx.x;
  if (idx < K * N) {
    int k = idx / N, n = idx - k * N;
    Wt[(size_t)n * K + k] = (bf16_t)W[idx];
  }
}

// ---------------------------------------------------------------------------
// LayerNorm over 512 channels -> bf16 (one block per token row)
// ---------------------------------------------------------------------------
__global__ void ln512_kernel(const float* __restrict__ x, const float* __restrict__ g,
                             const float* __restrict__ be, bf16_t* __restrict__ y) {
  __shared__ float s1[256];
  __shared__ float s2[256];
  int row = blockIdx.x, tid = threadIdx.x;
  const float* xr = x + (size_t)row * DMODEL;
  float a0 = xr[tid], a1 = xr[tid + 256];
  s1[tid] = a0 + a1;
  s2[tid] = a0 * a0 + a1 * a1;
  __syncthreads();
  for (int off = 128; off > 0; off >>= 1) {
    if (tid < off) { s1[tid] += s1[tid + off]; s2[tid] += s2[tid + off]; }
    __syncthreads();
  }
  float mu  = s1[0] * (1.0f / DMODEL);
  float var = s2[0] * (1.0f / DMODEL) - mu * mu;
  float inv = rsqrtf(var + 1e-5f);
  bf16_t* yr = y + (size_t)row * DMODEL;
  yr[tid]       = (bf16_t)((a0 - mu) * inv * g[tid]       + be[tid]);
  yr[tid + 256] = (bf16_t)((a1 - mu) * inv * g[tid + 256] + be[tid + 256]);
}

// ---------------------------------------------------------------------------
// q/k projection: [8192x512] @ Wt[512][512]; scatter into [b][head][tok][dk]
// head split per reference view(b,n,dk,h): c = dk*8 + head
// block tile 64x256 (2 waves along M, 4 along N); grid (2, 128)
// ---------------------------------------------------------------------------
__global__ void proj_qk_kernel(const bf16_t* __restrict__ A, const bf16_t* __restrict__ Bt,
                               const float* __restrict__ bias, bf16_t* __restrict__ outH) {
  int lane = threadIdx.x & 31, wave = threadIdx.x >> 5;
  int m0 = blockIdx.y * 64 + (wave & 1) * 32;
  int n0 = blockIdx.x * 256 + (wave >> 1) * 64;
  ZERO_ACC(acc);
  gemm_tile(A, A, DMODEL, DMODEL, DMODEL, Bt, DMODEL, m0, n0, lane, acc);
#pragma unroll
  for (int p = 0; p < 2; ++p) {
#pragma unroll
    for (int t = 0; t < 4; ++t) {
      int c = n0 + t * 16 + (lane & 15);
      float bv = bias[c];
      int head = c & 7, dki = c >> 3;
#pragma unroll
      for (int i = 0; i < 8; ++i) {
        int m = m0 + p * 16 + ((lane >> 4) << 3) + i;
        int bat = m >> 10, tok = m & 1023;
        outH[(((size_t)bat * NHEAD + head) * NTOK + tok) * DK + dki] =
            (bf16_t)(acc[p][t][i] + bv);
      }
    }
  }
}

// v projection: concat([r,d]) [8192x1024] @ Wt[512][1024]; scatter transposed
// into [b][head][dk][tok] so AV B-fragments are contiguous in K=j.
__global__ void proj_v_kernel(const bf16_t* __restrict__ A1, const bf16_t* __restrict__ A2,
                              const bf16_t* __restrict__ Bt, const float* __restrict__ bias,
                              bf16_t* __restrict__ outT) {
  int lane = threadIdx.x & 31, wave = threadIdx.x >> 5;
  int m0 = blockIdx.y * 64 + (wave & 1) * 32;
  int n0 = blockIdx.x * 256 + (wave >> 1) * 64;
  ZERO_ACC(acc);
  gemm_tile(A1, A2, DMODEL, 2 * DMODEL, DMODEL, Bt, 2 * DMODEL, m0, n0, lane, acc);
#pragma unroll
  for (int p = 0; p < 2; ++p) {
#pragma unroll
    for (int t = 0; t < 4; ++t) {
      int c = n0 + t * 16 + (lane & 15);
      float bv = bias[c];
      int head = c & 7, dki = c >> 3;
#pragma unroll
      for (int i = 0; i < 8; ++i) {
        int m = m0 + p * 16 + ((lane >> 4) << 3) + i;
        int bat = m >> 10, tok = m & 1023;
        outT[(((size_t)bat * NHEAD + head) * DK + dki) * NTOK + tok] =
            (bf16_t)(acc[p][t][i] + bv);
      }
    }
  }
}

// ---------------------------------------------------------------------------
// scores = q @ k^T per (b,h), K=dk=64; writes S and S^T (S^T feeds col-softmax)
// grid: (4, 16, chunk)
// ---------------------------------------------------------------------------
__global__ void scores_kernel(const bf16_t* __restrict__ qh, const bf16_t* __restrict__ kh,
                              float* __restrict__ s, float* __restrict__ st, int bh0) {
  int z = blockIdx.z, bh = bh0 + z;
  const bf16_t* q = qh + (size_t)bh * NTOK * DK;
  const bf16_t* k = kh + (size_t)bh * NTOK * DK;
  float* sz  = s  + (size_t)z * NTOK * NTOK;
  float* stz = st + (size_t)z * NTOK * NTOK;
  int lane = threadIdx.x & 31, wave = threadIdx.x >> 5;
  int m0 = blockIdx.y * 64 + (wave & 1) * 32;
  int n0 = blockIdx.x * 256 + (wave >> 1) * 64;
  ZERO_ACC(acc);
  gemm_tile(q, q, DK, DK, DK, k, DK, m0, n0, lane, acc);
#pragma unroll
  for (int p = 0; p < 2; ++p) {
#pragma unroll
    for (int t = 0; t < 4; ++t) {
      int n = n0 + t * 16 + (lane & 15);
#pragma unroll
      for (int i = 0; i < 8; ++i) {
        int m = m0 + p * 16 + ((lane >> 4) << 3) + i;
        float v = acc[p][t][i];
        sz [(size_t)m * NTOK + n] = v;
        stz[(size_t)n * NTOK + m] = v;
      }
    }
  }
}

// softmax over a 1024-wide row -> bf16 (serves row-softmax on S, col-softmax on S^T)
// grid: (1024, chunk)
__global__ void softmax_row_kernel(const float* __restrict__ s, bf16_t* __restrict__ out) {
  __shared__ float red[256];
  size_t base = (size_t)blockIdx.y * NTOK * NTOK + (size_t)blockIdx.x * NTOK;
  int tid = threadIdx.x;
  const float* sr = s + base;
  float v[4];
  float mx = -3.0e38f;
#pragma unroll
  for (int i = 0; i < 4; ++i) { v[i] = sr[tid + i * 256]; mx = fmaxf(mx, v[i]); }
  red[tid] = mx; __syncthreads();
  for (int off = 128; off > 0; off >>= 1) {
    if (tid < off) red[tid] = fmaxf(red[tid], red[tid + off]);
    __syncthreads();
  }
  mx = red[0]; __syncthreads();
  float sum = 0.f;
#pragma unroll
  for (int i = 0; i < 4; ++i) { v[i] = __expf(v[i] - mx); sum += v[i]; }
  red[tid] = sum; __syncthreads();
  for (int off = 128; off > 0; off >>= 1) {
    if (tid < off) red[tid] += red[tid + off];
    __syncthreads();
  }
  float inv = 1.0f / red[0];
  bf16_t* orow = out + base;
#pragma unroll
  for (int i = 0; i < 4; ++i) orow[tid + i * 256] = (bf16_t)(v[i] * inv);
}

// result = att @ v  (M=1024, N=64, K=1024), fused residual add into d_out f32.
// 8 waves along M, 32 rows each -> block covers 256 rows; grid: (4, chunk)
__global__ void av_kernel(const bf16_t* __restrict__ att, const bf16_t* __restrict__ vt,
                          const float* __restrict__ residual, float* __restrict__ outp,
                          int bh0) {
  int z = blockIdx.y, bh = bh0 + z;
  int b = bh >> 3, h = bh & 7;
  const bf16_t* a  = att + (size_t)z  * NTOK * NTOK;
  const bf16_t* vz = vt  + (size_t)bh * DK * NTOK;
  int lane = threadIdx.x & 31, wave = threadIdx.x >> 5;
  int m0 = blockIdx.x * 256 + wave * 32;
  ZERO_ACC(acc);
  gemm_tile(a, a, NTOK, NTOK, NTOK, vz, NTOK, m0, 0, lane, acc);
#pragma unroll
  for (int p = 0; p < 2; ++p) {
#pragma unroll
    for (int t = 0; t < 4; ++t) {
      int d = t * 16 + (lane & 15);
      int c = d * 8 + h;
#pragma unroll
      for (int i = 0; i < 8; ++i) {
        int m = m0 + p * 16 + ((lane >> 4) << 3) + i;
        size_t idx = ((size_t)b * NTOK + m) * DMODEL + c;
        outp[idx] = residual[idx] + acc[p][t][i];
      }
    }
  }
}

// FFN layer 1: [8192x512] @ Wt[1024][512], relu -> bf16 mid; grid (4, 128)
__global__ void ffn1_kernel(const bf16_t* __restrict__ A, const bf16_t* __restrict__ Bt,
                            const float* __restrict__ bias, bf16_t* __restrict__ mid) {
  int lane = threadIdx.x & 31, wave = threadIdx.x >> 5;
  int m0 = blockIdx.y * 64 + (wave & 1) * 32;
  int n0 = blockIdx.x * 256 + (wave >> 1) * 64;
  ZERO_ACC(acc);
  gemm_tile(A, A, DMODEL, DMODEL, DMODEL, Bt, DMODEL, m0, n0, lane, acc);
#pragma unroll
  for (int p = 0; p < 2; ++p) {
#pragma unroll
    for (int t = 0; t < 4; ++t) {
      int c = n0 + t * 16 + (lane & 15);
      float bv = bias[c];
#pragma unroll
      for (int i = 0; i < 8; ++i) {
        int m = m0 + p * 16 + ((lane >> 4) << 3) + i;
        mid[(size_t)m * (2 * DMODEL) + c] = (bf16_t)fmaxf(acc[p][t][i] + bv, 0.f);
      }
    }
  }
}

// FFN layer 2: [8192x1024] @ Wt[512][1024], fused second residual; grid (2, 128)
__global__ void ffn2_kernel(const bf16_t* __restrict__ A, const bf16_t* __restrict__ Bt,
                            const float* __restrict__ bias, float* __restrict__ outp) {
  int lane = threadIdx.x & 31, wave = threadIdx.x >> 5;
  int m0 = blockIdx.y * 64 + (wave & 1) * 32;
  int n0 = blockIdx.x * 256 + (wave >> 1) * 64;
  ZERO_ACC(acc);
  gemm_tile(A, A, 2 * DMODEL, 2 * DMODEL, 2 * DMODEL, Bt, 2 * DMODEL, m0, n0, lane, acc);
#pragma unroll
  for (int p = 0; p < 2; ++p) {
#pragma unroll
    for (int t = 0; t < 4; ++t) {
      int c = n0 + t * 16 + (lane & 15);
      float bv = bias[c];
#pragma unroll
      for (int i = 0; i < 8; ++i) {
        int m = m0 + p * 16 + ((lane >> 4) << 3) + i;
        size_t idx = (size_t)m * DMODEL + c;
        outp[idx] = outp[idx] + acc[p][t][i] + bv;
      }
    }
  }
}

__global__ void copy_kernel(const float* __restrict__ src, float* __restrict__ dst, int n) {
  int i = blockIdx.x * blockDim.x + threadIdx.x;
  int stride = gridDim.x * blockDim.x;
  for (; i < n; i += stride) dst[i] = src[i];
}

// ---------------------------------------------------------------------------
// Host orchestration
// ---------------------------------------------------------------------------
extern "C" void kernel_launch(void* const* d_in, const int* in_sizes, int n_in,
                              void* d_out, int out_size, void* d_ws, size_t ws_size,
                              hipStream_t stream) {
  (void)in_sizes; (void)n_in; (void)out_size;
  const float* rgb1   = (const float*)d_in[0];
  const float* depth1 = (const float*)d_in[1];
  const float* Wq  = (const float*)d_in[2];  const float* bq  = (const float*)d_in[3];
  const float* Wk  = (const float*)d_in[4];  const float* bk  = (const float*)d_in[5];
  const float* Wv1 = (const float*)d_in[6];  const float* bv1 = (const float*)d_in[7];
  const float* Wv2 = (const float*)d_in[8];  const float* bv2 = (const float*)d_in[9];
  const float* W11 = (const float*)d_in[10]; const float* b11 = (const float*)d_in[11];
  const float* W12 = (const float*)d_in[12]; const float* b12 = (const float*)d_in[13];
  const float* W21 = (const float*)d_in[14]; const float* b21 = (const float*)d_in[15];
  const float* W22 = (const float*)d_in[16]; const float* b22 = (const float*)d_in[17];
  const float* g11 = (const float*)d_in[18]; const float* be11 = (const float*)d_in[19];
  const float* g12 = (const float*)d_in[20]; const float* be12 = (const float*)d_in[21];
  const float* g21 = (const float*)d_in[22]; const float* be21 = (const float*)d_in[23];
  const float* g22 = (const float*)d_in[24]; const float* be22 = (const float*)d_in[25];

  // --- workspace carve-up (256B aligned) ---
  char* ws = (char*)d_ws;
  size_t off = 0;
  auto carve = [&](size_t bytes) {
    size_t o = (off + 255) & ~(size_t)255; off = o + bytes; return o;
  };
  const size_t ACT_BF = (size_t)ROWS * DMODEL * sizeof(bf16_t);      // 8 MiB
  bf16_t* r16  = (bf16_t*)(ws + carve(ACT_BF));
  bf16_t* d16  = (bf16_t*)(ws + carve(ACT_BF));
  bf16_t* qh   = (bf16_t*)(ws + carve(ACT_BF));   // [b][h][tok][dk]
  bf16_t* kh   = (bf16_t*)(ws + carve(ACT_BF));
  bf16_t* v1t  = (bf16_t*)(ws + carve(ACT_BF));   // [b][h][dk][tok]
  bf16_t* v2t  = (bf16_t*)(ws + carve(ACT_BF));
  bf16_t* WqT  = (bf16_t*)(ws + carve((size_t)512 * 512 * 2));
  bf16_t* WkT  = (bf16_t*)(ws + carve((size_t)512 * 512 * 2));
  bf16_t* Wv1T = (bf16_t*)(ws + carve((size_t)512 * 1024 * 2));
  bf16_t* Wv2T = (bf16_t*)(ws + carve((size_t)512 * 1024 * 2));
  bf16_t* W11T = (bf16_t*)(ws + carve((size_t)1024 * 512 * 2));
  bf16_t* W12T = (bf16_t*)(ws + carve((size_t)512 * 1024 * 2));
  bf16_t* W21T = (bf16_t*)(ws + carve((size_t)1024 * 512 * 2));
  bf16_t* W22T = (bf16_t*)(ws + carve((size_t)512 * 1024 * 2));
  bf16_t* ln16 = (bf16_t*)(ws + carve(ACT_BF));
  bf16_t* mid16= (bf16_t*)(ws + carve((size_t)ROWS * 2 * DMODEL * 2));

  // Adaptive attention chunking: per (b,h) pair needs S(f32)+S^T(f32)+attA+attB
  const size_t NN = (size_t)NTOK * NTOK;
  const size_t per_bh = 2 * NN * 4 + 2 * NN * 2 + 4 * 256;           // ~12.6 MB
  size_t fixed_end = (off + 255) & ~(size_t)255;
  size_t avail = ws_size > fixed_end ? ws_size - fixed_end : 0;
  int chunk = 1;
  while (chunk < BH && (size_t)(chunk * 2) * per_bh <= avail) chunk *= 2;
  float*  scores  = (float*) (ws + carve(NN * 4 * chunk));
  float*  scoresT = (float*) (ws + carve(NN * 4 * chunk));
  bf16_t* attA    = (bf16_t*)(ws + carve(NN * 2 * chunk));
  bf16_t* attB    = (bf16_t*)(ws + carve(NN * 2 * chunk));

  float* out_r  = (float*)d_out;
  float* out_d  = out_r + (size_t)ROWS * DMODEL;
  float* out_d2 = out_d + (size_t)ROWS * DMODEL;

  dim3 blk(256);

  // 1) weights -> bf16 transposed [out][in]
  wt_kernel<<<(512 * 512  + 255) / 256, blk, 0, stream>>>(Wq,  WqT,  512,  512);
  wt_kernel<<<(512 * 512  + 255) / 256, blk, 0, stream>>>(Wk,  WkT,  512,  512);
  wt_kernel<<<(1024 * 512 + 255) / 256, blk, 0, stream>>>(Wv1, Wv1T, 1024, 512);
  wt_kernel<<<(1024 * 512 + 255) / 256, blk, 0, stream>>>(Wv2, Wv2T, 1024, 512);
  wt_kernel<<<(512 * 1024 + 255) / 256, blk, 0, stream>>>(W11, W11T, 512,  1024);
  wt_kernel<<<(1024 * 512 + 255) / 256, blk, 0, stream>>>(W12, W12T, 1024, 512);
  wt_kernel<<<(512 * 1024 + 255) / 256, blk, 0, stream>>>(W21, W21T, 512,  1024);
  wt_kernel<<<(1024 * 512 + 255) / 256, blk, 0, stream>>>(W22, W22T, 1024, 512);

  // 2) pre-norm both streams -> bf16
  ln512_kernel<<<ROWS, blk, 0, stream>>>(rgb1,   g11, be11, r16);
  ln512_kernel<<<ROWS, blk, 0, stream>>>(depth1, g12, be12, d16);

  // 3) projections (WMMA)
  dim3 gproj(2, 128);
  proj_qk_kernel<<<gproj, blk, 0, stream>>>(r16, WqT, bq, qh);
  proj_qk_kernel<<<gproj, blk, 0, stream>>>(d16, WkT, bk, kh);
  proj_v_kernel <<<gproj, blk, 0, stream>>>(r16, d16, Wv1T, bv1, v1t);
  proj_v_kernel <<<gproj, blk, 0, stream>>>(r16, d16, Wv2T, bv2, v2t);

  // 4) attention, chunked over (b,h)
  for (int bh0 = 0; bh0 < BH; bh0 += chunk) {
    scores_kernel<<<dim3(4, 16, chunk), blk, 0, stream>>>(qh, kh, scores, scoresT, bh0);
    softmax_row_kernel<<<dim3(NTOK, chunk), blk, 0, stream>>>(scores,  attA);
    softmax_row_kernel<<<dim3(NTOK, chunk), blk, 0, stream>>>(scoresT, attB);
    av_kernel<<<dim3(4, chunk), blk, 0, stream>>>(attA, v1t, rgb1,   out_r, bh0);
    av_kernel<<<dim3(4, chunk), blk, 0, stream>>>(attB, v2t, depth1, out_d, bh0);
  }

  // 5) FFN rgb: res1_r = res_r + W12(relu(W11(LN(res_r))))
  ln512_kernel<<<ROWS, blk, 0, stream>>>(out_r, g21, be21, ln16);
  ffn1_kernel<<<dim3(4, 128), blk, 0, stream>>>(ln16, W11T, b11, mid16);
  ffn2_kernel<<<dim3(2, 128), blk, 0, stream>>>(mid16, W12T, b12, out_r);

  // 6) FFN depth
  ln512_kernel<<<ROWS, blk, 0, stream>>>(out_d, g22, be22, ln16);
  ffn1_kernel<<<dim3(4, 128), blk, 0, stream>>>(ln16, W21T, b21, mid16);
  ffn2_kernel<<<dim3(2, 128), blk, 0, stream>>>(mid16, W22T, b22, out_d);

  // 7) third output duplicates res1_d
  copy_kernel<<<4096, blk, 0, stream>>>(out_d, out_d2, ROWS * DMODEL);
}
